// OT_Attn_15857019256980
// MI455X (gfx1250) — matrix operand order repaired and checked by reading, not compiled, not run on hardware
//
#include <hip/hip_runtime.h>
#include <cstdint>
#include <cstddef>

typedef __bf16 bf16;
typedef __attribute__((ext_vector_type(16))) __bf16 v16bf;
typedef __attribute__((ext_vector_type(8)))  __bf16 v8bf;
typedef __attribute__((ext_vector_type(8)))  float  v8f;

constexpr int    N  = 8192;
constexpr int    M  = 2048;
constexpr int    K  = 1024;
constexpr size_t NM = (size_t)N * (size_t)M;

constexpr float EPS      = 0.1f;
constexpr float INV_EPS  = 10.0f;
constexpr float LAM_EPS  = 0.1f / 1.1f;          // eps * gamma/(gamma+eps)
constexpr float LOGA     = -9.010913347f;        // -log(8192)
constexpr float LOG_M    = 7.624618986f;         // log(2048)

#define DEVFN __device__ __forceinline__

// ---------------------------------------------------------------- scalars
__global__ void k_init(const int* it_p, const int* ipe_p, float* scal) {
  if (threadIdx.x == 0) {
    float total = 10.0f * (float)ipe_p[0];
    float ramp;
    if (total <= 0.0f) {
      ramp = 1.0f;
    } else {
      float cur   = fminf(fmaxf((float)it_p[0], 0.0f), total);
      float phase = 1.0f - cur / total;
      ramp = __expf(-5.0f * phase * phase);
    }
    scal[0] = fminf(ramp * 1.0f + 0.1f, 1.0f);   // rho
    ((unsigned*)scal)[1] = 0u;                   // max(cost) accumulator (uint bits)
  }
}

__global__ void k_zero(float* p, int n) {
  int i = blockIdx.x * 256 + threadIdx.x;
  if (i < n) p[i] = 0.0f;
}

// ------------------------------------------- row-min normalize + bf16 split
// one block per row of 1024 floats
__global__ void __launch_bounds__(256)
k_norm(const float* __restrict__ src, bf16* __restrict__ hi, bf16* __restrict__ lo,
       float* __restrict__ sq) {
  const int row = blockIdx.x;
  const int t   = threadIdx.x;
  const float* p = src + (size_t)row * K;
  float v[4];
  float mn = INFINITY;
#pragma unroll
  for (int c = 0; c < 4; ++c) { v[c] = p[t + c * 256]; mn = fminf(mn, v[c]); }

  __shared__ float sm[256];
  sm[t] = mn; __syncthreads();
  for (int off = 128; off > 0; off >>= 1) {
    if (t < off) sm[t] = fminf(sm[t], sm[t + off]);
    __syncthreads();
  }
  mn = sm[0];
  __syncthreads();

  float ssq = 0.0f;
#pragma unroll
  for (int c = 0; c < 4; ++c) {
    float x = v[c] - mn;
    bf16 h = (bf16)x;
    bf16 l = (bf16)(x - (float)h);
    size_t idx = (size_t)row * K + t + c * 256;
    hi[idx] = h;
    lo[idx] = l;
    ssq += x * x;
  }
  sm[t] = ssq; __syncthreads();
  for (int off = 128; off > 0; off >>= 1) {
    if (t < off) sm[t] += sm[t + off];
    __syncthreads();
  }
  if (t == 0) sq[row] = sm[0];
}

// ---------------------------------------------------------------- WMMA GEMM
DEVFN v8f wmma_bf16(v16bf a, v16bf b, v8f c) {
  return __builtin_amdgcn_wmma_f32_16x16x32_bf16(
      /*neg_a=*/false, a, /*neg_b=*/false, b,
      /*c_mod=*/(short)0, c, /*reuse_a=*/false, /*reuse_b=*/false);
}

// A tile element layout (16x32 bf16): lane (r, half): elems 0..7 = K ka..ka+7,
// elems 8..15 = K ka+16..ka+23 where ka = k0 + half*8
DEVFN v16bf load_a(const bf16* base, int row, int ka) {
  const bf16* p = base + (size_t)row * K + ka;
  v8bf c0 = *(const v8bf*)p;
  v8bf c1 = *(const v8bf*)(p + 16);
  return __builtin_shufflevector(c0, c1, 0,1,2,3,4,5,6,7,8,9,10,11,12,13,14,15);
}

// B tile (32x16 bf16), column = y row (row-major y gives contiguous K):
// lane (r, half): elems 0..7 = K kb..kb+7, elems 8..15 = K kb+16..kb+23, kb = k0 + half*8
DEVFN v16bf load_b(const bf16* base, int col, int kb) {
  const bf16* p = base + (size_t)col * K + kb;
  v8bf c0 = *(const v8bf*)p;
  v8bf c1 = *(const v8bf*)(p + 16);
  return __builtin_shufflevector(c0, c1, 0,1,2,3,4,5,6,7,8,9,10,11,12,13,14,15);
}

__global__ void __launch_bounds__(256)
k_gemm_cost(const bf16* __restrict__ xhi, const bf16* __restrict__ xlo,
            const bf16* __restrict__ yhi, const bf16* __restrict__ ylo,
            const float* __restrict__ x2, const float* __restrict__ y2,
            float* __restrict__ cost) {
  const int w    = threadIdx.x >> 5;
  const int lane = threadIdx.x & 31;
  const int r    = lane & 15;
  const int half = lane >> 4;
  const int row0 = blockIdx.y * 128 + (w & 3) * 32;   // 4 wave-rows
  const int col0 = blockIdx.x * 64  + (w >> 2) * 32;  // 2 wave-cols

  v8f acc[2][2] = {};

  for (int k0 = 0; k0 < K; k0 += 32) {
    const int ka = k0 + half * 8;
    const int kb = k0 + half * 8;
    v16bf ah[2], al[2], bh[2], bl[2];
#pragma unroll
    for (int mi = 0; mi < 2; ++mi) {
      ah[mi] = load_a(xhi, row0 + mi * 16 + r, ka);
      al[mi] = load_a(xlo, row0 + mi * 16 + r, ka);
    }
#pragma unroll
    for (int ni = 0; ni < 2; ++ni) {
      bh[ni] = load_b(yhi, col0 + ni * 16 + r, kb);
      bl[ni] = load_b(ylo, col0 + ni * 16 + r, kb);
    }
#pragma unroll
    for (int mi = 0; mi < 2; ++mi) {
#pragma unroll
      for (int ni = 0; ni < 2; ++ni) {
        acc[mi][ni] = wmma_bf16(ah[mi], bh[ni], acc[mi][ni]);  // hi*hi
        acc[mi][ni] = wmma_bf16(ah[mi], bl[ni], acc[mi][ni]);  // hi*lo
        acc[mi][ni] = wmma_bf16(al[mi], bh[ni], acc[mi][ni]);  // lo*hi
      }
    }
  }

  // C/D layout: lanes 0-15 N=lane, VGPR p -> M=p (+8 for lanes 16-31)
#pragma unroll
  for (int mi = 0; mi < 2; ++mi) {
#pragma unroll
    for (int p = 0; p < 8; ++p) {
      const int rr = row0 + mi * 16 + half * 8 + p;
      const float xs = x2[rr];
#pragma unroll
      for (int ni = 0; ni < 2; ++ni) {
        const int cc = col0 + ni * 16 + r;
        float d2 = xs + y2[cc] - 2.0f * acc[mi][ni][p];
        cost[(size_t)rr * M + cc] = sqrtf(fmaxf(d2, 0.0f));
      }
    }
  }
}

// ---------------------------------------------------------------- max(cost)
__global__ void __launch_bounds__(256)
k_max(const float* __restrict__ cost, unsigned* __restrict__ maxbits) {
  const size_t stride = (size_t)gridDim.x * 256;
  float m = 0.0f;  // cost >= 0
  for (size_t idx = (size_t)blockIdx.x * 256 + threadIdx.x; idx < NM; idx += stride)
    m = fmaxf(m, cost[idx]);
  __shared__ float sm[256];
  sm[threadIdx.x] = m; __syncthreads();
  for (int off = 128; off > 0; off >>= 1) {
    if (threadIdx.x < off) sm[threadIdx.x] = fmaxf(sm[threadIdx.x], sm[threadIdx.x + off]);
    __syncthreads();
  }
  if (threadIdx.x == 0) atomicMax(maxbits, __float_as_uint(sm[0]));
}

// ---------------------------------------------------------------- LSE utils
DEVFN void lse_acc(float& m, float& s, float v) {
  float nm = fmaxf(m, v);
  s = s * __expf(m - nm) + __expf(v - nm);
  m = nm;
}
DEVFN void lse_merge(float& m, float& s, float m2, float s2) {
  float nm = fmaxf(m, m2);
  s = s * __expf(m - nm) + s2 * __expf(m2 - nm);
  m = nm;
}

// f_i = -lam*eps * LSE_j( -cn_ij/eps + g_j/eps + logb ); one block per row
__global__ void __launch_bounds__(256)
k_frow(const float* __restrict__ cost, const float* __restrict__ g,
       float* __restrict__ f, const float* __restrict__ scal) {
  const int i = blockIdx.x;
  const float scaleK = INV_EPS / scal[1];          // 1/(maxc*eps)
  const float logb   = __logf(scal[0]) - LOG_M;
  const float* row = cost + (size_t)i * M;
  float m = -INFINITY, s = 0.0f;
  for (int j = threadIdx.x; j < M; j += 256)
    lse_acc(m, s, -row[j] * scaleK + g[j] * INV_EPS + logb);
  __shared__ float sm[256], ss[256];
  sm[threadIdx.x] = m; ss[threadIdx.x] = s; __syncthreads();
  for (int off = 128; off > 0; off >>= 1) {
    if (threadIdx.x < off) {
      float mm = sm[threadIdx.x], s0 = ss[threadIdx.x];
      lse_merge(mm, s0, sm[threadIdx.x + off], ss[threadIdx.x + off]);
      sm[threadIdx.x] = mm; ss[threadIdx.x] = s0;
    }
    __syncthreads();
  }
  if (threadIdx.x == 0) f[i] = -LAM_EPS * (sm[0] + __logf(ss[0]));
}

// g_j = -eps * LSE_i( -cn_ij/eps + f_i/eps + loga ); 32 columns per block
__global__ void __launch_bounds__(256)
k_gcol(const float* __restrict__ cost, const float* __restrict__ f,
       float* __restrict__ g, const float* __restrict__ scal) {
  const int tx = threadIdx.x & 31;
  const int ty = threadIdx.x >> 5;                 // 0..7
  const int j  = blockIdx.x * 32 + tx;
  const float scaleK = INV_EPS / scal[1];
  float m = -INFINITY, s = 0.0f;
  for (int i = ty; i < N; i += 8)
    lse_acc(m, s, -cost[(size_t)i * M + j] * scaleK + f[i] * INV_EPS + LOGA);
  __shared__ float sm[256], ss[256];
  sm[threadIdx.x] = m; ss[threadIdx.x] = s; __syncthreads();
  if (ty == 0) {
    for (int k = 1; k < 8; ++k) lse_merge(m, s, sm[k * 32 + tx], ss[k * 32 + tx]);
    g[j] = -EPS * (m + __logf(s));
  }
}

// flow = exp(logK + f/eps + g/eps + loga + logb) / rho  (in-place over cost),
// per-block partial of sum(cn*flow) for dist
__global__ void __launch_bounds__(256)
k_flow(float* __restrict__ cf, const float* __restrict__ f, const float* __restrict__ g,
       const float* __restrict__ scal, float* __restrict__ part) {
  const float rho    = scal[0];
  const float invmax = 1.0f / scal[1];
  const float logb   = __logf(rho) - LOG_M;
  const float invrho = 1.0f / rho;
  const size_t stride = (size_t)gridDim.x * 256;
  float acc = 0.0f;
  for (size_t idx = (size_t)blockIdx.x * 256 + threadIdx.x; idx < NM; idx += stride) {
    const int i = (int)(idx >> 11);
    const int j = (int)(idx & 2047);
    float cn = cf[idx] * invmax;
    float e  = (-cn + f[i] + g[j]) * INV_EPS + LOGA + logb;
    float fl = __expf(e) * invrho;
    cf[idx] = fl;
    acc += cn * fl;
  }
  __shared__ float sm[256];
  sm[threadIdx.x] = acc; __syncthreads();
  for (int off = 128; off > 0; off >>= 1) {
    if (threadIdx.x < off) sm[threadIdx.x] += sm[threadIdx.x + off];
    __syncthreads();
  }
  if (threadIdx.x == 0) part[blockIdx.x] = sm[0];
}

__global__ void __launch_bounds__(256)
k_dist(const float* __restrict__ part, float* __restrict__ out) {
  float s = 0.0f;
  for (int k = threadIdx.x; k < 4096; k += 256) s += part[k];
  __shared__ float sm[256];
  sm[threadIdx.x] = s; __syncthreads();
  for (int off = 128; off > 0; off >>= 1) {
    if (threadIdx.x < off) sm[threadIdx.x] += sm[threadIdx.x + off];
    __syncthreads();
  }
  if (threadIdx.x == 0) out[0] = sm[0];
}

// ---------------------------------------------------------------- launcher
extern "C" void kernel_launch(void* const* d_in, const int* in_sizes, int n_in,
                              void* d_out, int out_size, void* d_ws, size_t ws_size,
                              hipStream_t stream) {
  const float* x   = (const float*)d_in[0];
  const float* y   = (const float*)d_in[1];
  const int*   it  = (const int*)d_in[2];
  const int*   ipe = (const int*)d_in[3];
  float* out = (float*)d_out;                 // [NM] flow (+cost scratch), [1] dist

  char* w = (char*)d_ws;
  bf16* xhi = (bf16*)w; w += (size_t)N * K * 2;
  bf16* xlo = (bf16*)w; w += (size_t)N * K * 2;
  bf16* yhi = (bf16*)w; w += (size_t)M * K * 2;
  bf16* ylo = (bf16*)w; w += (size_t)M * K * 2;
  float* x2 = (float*)w; w += (size_t)N * 4;
  float* y2 = (float*)w; w += (size_t)M * 4;
  float* fv = (float*)w; w += (size_t)N * 4;
  float* gv = (float*)w; w += (size_t)M * 4;
  float* scal = (float*)w; w += 256;
  float* part = (float*)w; w += 4096 * 4;

  k_init<<<1, 32, 0, stream>>>(it, ipe, scal);
  k_zero<<<8, 256, 0, stream>>>(gv, M);
  k_norm<<<N, 256, 0, stream>>>(x, xhi, xlo, x2);
  k_norm<<<M, 256, 0, stream>>>(y, yhi, ylo, y2);
  k_gemm_cost<<<dim3(M / 64, N / 128), 256, 0, stream>>>(xhi, xlo, yhi, ylo, x2, y2, out);
  k_max<<<4096, 256, 0, stream>>>(out, ((unsigned*)scal) + 1);
  for (int s = 0; s < 3; ++s) {
    k_frow<<<N, 256, 0, stream>>>(out, gv, fv, scal);
    k_gcol<<<M / 32, 256, 0, stream>>>(out, fv, gv, scal);
  }
  k_flow<<<4096, 256, 0, stream>>>(out, fv, gv, scal, part);
  k_dist<<<1, 256, 0, stream>>>(part, out + NM);
}
